// MambaIR_46291157517012
// MI455X (gfx1250) — matrix operand order repaired and checked by reading, not compiled, not run on hardware
//
#include <hip/hip_runtime.h>
#include <hip/hip_bf16.h>
#include <math.h>

// ---- fixed problem dimensions (from setup_inputs) ----
#define L_    1728      // 12*12*12
#define C_    96
#define DI_   192
#define K6_   6
#define NS_   16
#define R_    6
#define RN_   38        // R + 2N
#define RP_   48        // RN_ padded to a multiple of 16

typedef __attribute__((ext_vector_type(16))) _Float16 v16h;
typedef __attribute__((ext_vector_type(8)))  float    v8f;

__device__ __forceinline__ float silu_f(float x) { return x / (1.0f + __expf(-x)); }

// ============================================================================
// WMMA GEMM, all dims tile-exact (no bounds checks):
//   C[m][n] = sum_k A[m*lda + k] * B[n*ldb + k]        (both K-contiguous)
// One wave per 16x16 tile. K is a compile-time constant -> fully unrolled
// chain of global_load_b128 -> v_cvt_f16 -> v_wmma_f32_16x16x32_f16.
// Fragment layouts (ISA 7.12.2):
//   A: lane (hf,lrow) row=lrow, K runs [hf*8, hf*8+8) and [16+hf*8, ...)
//   B: lane (hf,lrow) col=lrow, K runs [hf*8, hf*8+8) and [16+hf*8, ...)
//   C/D: VGPR r -> row r (+8 for hf=1), col = lrow
// ============================================================================
template <int KD>
__global__ void gemm_wmma_kernel(const float* __restrict__ A,
                                 const float* __restrict__ B,
                                 float* __restrict__ Cm,
                                 int lda, int ldb, int ldc,
                                 long long batchA, long long batchB, long long batchC)
{
    const int lane = threadIdx.x & 31;
    const int hf   = lane >> 4;
    const int lrow = lane & 15;

    const float* Arow = A + (long long)blockIdx.z * batchA
                          + (long long)(blockIdx.x * 16 + lrow) * lda;
    const float* Brow = B + (long long)blockIdx.z * batchB
                          + (long long)(blockIdx.y * 16 + lrow) * ldb;

    v8f acc = {};
#pragma unroll
    for (int k0 = 0; k0 < KD; k0 += 32) {
        const float4* ap0 = (const float4*)(Arow + k0 + hf * 8);
        const float4* ap1 = (const float4*)(Arow + k0 + 16 + hf * 8);
        const float4* bp0 = (const float4*)(Brow + k0 + hf * 8);
        const float4* bp1 = (const float4*)(Brow + k0 + 16 + hf * 8);
        float4 a0 = ap0[0], a1 = ap0[1];
        float4 a2 = ap1[0], a3 = ap1[1];
        float4 b0 = bp0[0], b1 = bp0[1], b2 = bp1[0], b3 = bp1[1];

        v16h af, bf;
        af[0]  = (_Float16)a0.x; af[1]  = (_Float16)a0.y;
        af[2]  = (_Float16)a0.z; af[3]  = (_Float16)a0.w;
        af[4]  = (_Float16)a1.x; af[5]  = (_Float16)a1.y;
        af[6]  = (_Float16)a1.z; af[7]  = (_Float16)a1.w;
        af[8]  = (_Float16)a2.x; af[9]  = (_Float16)a2.y;
        af[10] = (_Float16)a2.z; af[11] = (_Float16)a2.w;
        af[12] = (_Float16)a3.x; af[13] = (_Float16)a3.y;
        af[14] = (_Float16)a3.z; af[15] = (_Float16)a3.w;

        bf[0]  = (_Float16)b0.x; bf[1]  = (_Float16)b0.y;
        bf[2]  = (_Float16)b0.z; bf[3]  = (_Float16)b0.w;
        bf[4]  = (_Float16)b1.x; bf[5]  = (_Float16)b1.y;
        bf[6]  = (_Float16)b1.z; bf[7]  = (_Float16)b1.w;
        bf[8]  = (_Float16)b2.x; bf[9]  = (_Float16)b2.y;
        bf[10] = (_Float16)b2.z; bf[11] = (_Float16)b2.w;
        bf[12] = (_Float16)b3.x; bf[13] = (_Float16)b3.y;
        bf[14] = (_Float16)b3.z; bf[15] = (_Float16)b3.w;

        acc = __builtin_amdgcn_wmma_f32_16x16x32_f16(false, af, false, bf,
                                                     (short)0, acc, false, false);
    }

    float* Crow = Cm + (long long)blockIdx.z * batchC
                     + (long long)(blockIdx.x * 16) * ldc
                     + blockIdx.y * 16 + lrow;
#pragma unroll
    for (int r = 0; r < 8; ++r)
        Crow[(long long)(hf ? r + 8 : r) * ldc] = acc[r];
}

// ============================================================================
// LayerNorm over last dim C (one thread per row), optional SiLU gate.
// ============================================================================
__global__ void ln_gate_kernel(const float* __restrict__ in,
                               const float* __restrict__ w,
                               const float* __restrict__ b,
                               const float* __restrict__ gate,
                               float* __restrict__ out,
                               int rows, int C, int in_stride,
                               int gate_stride, int gate_off, int out_stride)
{
    int r = blockIdx.x * blockDim.x + threadIdx.x;
    if (r >= rows) return;
    const float* x = in + (long long)r * in_stride;
    float mean = 0.0f;
    for (int c = 0; c < C; ++c) mean += x[c];
    mean /= (float)C;
    float var = 0.0f;
    for (int c = 0; c < C; ++c) { float d = x[c] - mean; var += d * d; }
    var /= (float)C;
    float inv = rsqrtf(var + 1e-5f);
    float* o = out + (long long)r * out_stride;
    for (int c = 0; c < C; ++c) {
        float v = (x[c] - mean) * inv * w[c] + b[c];
        if (gate) {
            float g = gate[(long long)r * gate_stride + gate_off + c];
            v *= silu_f(g);
        }
        o[c] = v;
    }
}

// ============================================================================
// Depthwise 3x3x3 conv (groups = Di) + SiLU. Output (L, Di) [coalesced].
// ============================================================================
__global__ void dwconv_silu_kernel(const float* __restrict__ xz,  // (L, 384)
                                   const float* __restrict__ w,   // (Di, 27)
                                   const float* __restrict__ bias,
                                   float* __restrict__ out)       // (L, Di)
{
    int idx = blockIdx.x * blockDim.x + threadIdx.x;
    if (idx >= L_ * DI_) return;
    int l = idx / DI_, c = idx % DI_;
    int d = l / 144, h = (l / 12) % 12, wq = l % 12;
    float s = bias[c];
    for (int kd = 0; kd < 3; ++kd) {
        int dd = d + kd - 1; if ((unsigned)dd >= 12u) continue;
        for (int kh = 0; kh < 3; ++kh) {
            int hh = h + kh - 1; if ((unsigned)hh >= 12u) continue;
            for (int kw = 0; kw < 3; ++kw) {
                int ww2 = wq + kw - 1; if ((unsigned)ww2 >= 12u) continue;
                int ln = dd * 144 + hh * 12 + ww2;
                s += w[c * 27 + kd * 9 + kh * 3 + kw] * xz[(long long)ln * (2 * DI_) + c];
            }
        }
    }
    out[(long long)l * DI_ + c] = silu_f(s);
}

// ============================================================================
// Build the 6 scan sequences in TRANSPOSED layout xs_t (6, L, Di):
//   xs_t[k][pos][c] = xm_t[src_k(pos)][c]   (K-contiguous for the GEMM,
//   lane-coalesced for scan/elementwise kernels)
// ============================================================================
__global__ void build_xs_kernel(const float* __restrict__ xm,   // (L, Di)
                                float* __restrict__ xs)         // (6, L, Di)
{
    int idx = blockIdx.x * blockDim.x + threadIdx.x;
    if (idx >= K6_ * L_ * DI_) return;
    int k   = idx / (L_ * DI_);
    int rem = idx % (L_ * DI_);
    int l   = rem / DI_;
    int c   = rem % DI_;
    int l2  = (k >= 3) ? (L_ - 1 - l) : l;
    int kb  = k % 3;
    int src;
    if (kb == 0) {
        src = l2;                                        // (d,h,w)
    } else if (kb == 1) {                                // order (h,w,d)
        int h = l2 / 144, w = (l2 / 12) % 12, d = l2 % 12;
        src = d * 144 + h * 12 + w;
    } else {                                             // order (w,d,h)
        int w = l2 / 144, d = (l2 / 12) % 12, h = l2 % 12;
        src = d * 144 + h * 12 + w;
    }
    xs[idx] = xm[(long long)src * DI_ + c];
}

// ============================================================================
// Pad x_proj_w (6, 38, 192) -> (6, 48, 192) with zero rows (tile-exact GEMM).
// ============================================================================
__global__ void pad_xproj_kernel(const float* __restrict__ w,
                                 float* __restrict__ wpad)
{
    int idx = blockIdx.x * blockDim.x + threadIdx.x;
    if (idx >= K6_ * RP_ * DI_) return;
    int b = idx / (RP_ * DI_);
    int m = (idx / DI_) % RP_;
    int k = idx % DI_;
    wpad[idx] = (m < RN_) ? w[((long long)b * RN_ + m) * DI_ + k] : 0.0f;
}

// ============================================================================
// delta_t[k][l][c] = softplus(sum_r dt_w[k][c][r] * xdbl[k][r][l] + dt_b[k][c])
// xdbl is the PADDED (6, 48, L) output of the x_proj GEMM.
// ============================================================================
__global__ void dt_kernel(const float* __restrict__ xdbl,   // (6, 48, L)
                          const float* __restrict__ dtw,    // (6, Di, 6)
                          const float* __restrict__ dtb,    // (6, Di)
                          float* __restrict__ delta)        // (6, L, Di)
{
    int idx = blockIdx.x * blockDim.x + threadIdx.x;
    if (idx >= K6_ * L_ * DI_) return;
    int k   = idx / (L_ * DI_);
    int rem = idx % (L_ * DI_);
    int l   = rem / DI_;
    int c   = rem % DI_;
    float s = dtb[k * DI_ + c];
    const float* wp = dtw + ((long long)k * DI_ + c) * R_;
    const float* xp = xdbl + (long long)k * RP_ * L_ + l;
#pragma unroll
    for (int r = 0; r < R_; ++r) s += wp[r] * xp[(long long)r * L_];
    delta[idx] = (s > 20.0f) ? s : log1pf(__expf(s));
}

// ============================================================================
// Selective scan: one thread per (k, c); 16 SSM states in registers,
// serial over L (exp on the TRANS pipe). Lane-coalesced (L, Di) layouts.
// ============================================================================
__global__ void scan_kernel(const float* __restrict__ xs,     // (6, L, Di)
                            const float* __restrict__ delta,  // (6, L, Di)
                            const float* __restrict__ xdbl,   // (6, 48, L)
                            const float* __restrict__ a_logs, // (6, Di, 16)
                            const float* __restrict__ dsv,    // (6, Di)
                            float* __restrict__ ys)           // (6, L, Di)
{
    int idx = blockIdx.x * blockDim.x + threadIdx.x;
    if (idx >= K6_ * DI_) return;
    int k = idx / DI_, c = idx % DI_;
    float Acoef[NS_], h[NS_];
#pragma unroll
    for (int n = 0; n < NS_; ++n) {
        Acoef[n] = -__expf(a_logs[((long long)k * DI_ + c) * NS_ + n]);
        h[n] = 0.0f;
    }
    float Dv = dsv[k * DI_ + c];
    const float* up = xs    + (long long)k * L_ * DI_ + c;
    const float* dp = delta + (long long)k * L_ * DI_ + c;
    const float* Bb = xdbl  + (long long)(k * RP_ + R_) * L_;
    const float* Cb = xdbl  + (long long)(k * RP_ + R_ + NS_) * L_;
    float* yp = ys + (long long)k * L_ * DI_ + c;
    for (int l = 0; l < L_; ++l) {
        float dl = dp[(long long)l * DI_];
        float u  = up[(long long)l * DI_];
        float du = dl * u;
        float acc = 0.0f;
#pragma unroll
        for (int n = 0; n < NS_; ++n) {
            float da = __expf(dl * Acoef[n]);
            h[n] = da * h[n] + du * Bb[(long long)n * L_ + l];
            acc += h[n] * Cb[(long long)n * L_ + l];
        }
        yp[(long long)l * DI_] = acc + Dv * u;
    }
}

// ============================================================================
// Merge 6 directional outputs back into (L, Di). Lane-coalesced.
// ============================================================================
__global__ void merge_kernel(const float* __restrict__ ys,  // (6, L, Di)
                             float* __restrict__ ym)        // (L, Di)
{
    int idx = blockIdx.x * blockDim.x + threadIdx.x;
    if (idx >= L_ * DI_) return;
    int l = idx / DI_, c = idx % DI_;
    int d = l / 144, h = (l / 12) % 12, w = l % 12;
    int l1 = h * 144 + w * 12 + d;   // position in (h,w,d) ordering
    int l2 = w * 144 + d * 12 + h;   // position in (w,d,h) ordering
    const long long ks = (long long)L_ * DI_;
    float v = ys[0 * ks + (long long)l  * DI_ + c] + ys[3 * ks + (long long)(L_ - 1 - l)  * DI_ + c]
            + ys[1 * ks + (long long)l1 * DI_ + c] + ys[4 * ks + (long long)(L_ - 1 - l1) * DI_ + c]
            + ys[2 * ks + (long long)l2 * DI_ + c] + ys[5 * ks + (long long)(L_ - 1 - l2) * DI_ + c];
    ym[(long long)l * DI_ + c] = v;
}

// ============================================================================
// out[i] = base[i] * skip[i % C] + add[i]
// ============================================================================
__global__ void res_kernel(const float* __restrict__ base,
                           const float* __restrict__ skip,
                           const float* __restrict__ add,
                           float* __restrict__ out, int n, int Cch)
{
    int i = blockIdx.x * blockDim.x + threadIdx.x;
    if (i >= n) return;
    out[i] = base[i] * skip[i % Cch] + add[i];
}

// ============================================================================
// Dense 3x3x3 conv, pad 1. Input element (ci, l) at in[l*si_l + ci*si_c].
// Output layout (Cout, L). act: 0 = none, 1 = exact GELU.
// ============================================================================
__global__ void conv3d_dense_kernel(const float* __restrict__ in,
                                    const float* __restrict__ w,   // (Cout, Cin, 27)
                                    const float* __restrict__ bias,
                                    float* __restrict__ out,
                                    int Cin, int Cout, int si_l, int si_c, int act)
{
    int idx = blockIdx.x * blockDim.x + threadIdx.x;
    if (idx >= Cout * L_) return;
    int o = idx / L_, l = idx % L_;
    int d = l / 144, h = (l / 12) % 12, wq = l % 12;
    float s = bias[o];
    for (int kd = 0; kd < 3; ++kd) {
        int dd = d + kd - 1; if ((unsigned)dd >= 12u) continue;
        for (int kh = 0; kh < 3; ++kh) {
            int hh = h + kh - 1; if ((unsigned)hh >= 12u) continue;
            for (int kw = 0; kw < 3; ++kw) {
                int ww2 = wq + kw - 1; if ((unsigned)ww2 >= 12u) continue;
                int ln = dd * 144 + hh * 12 + ww2;
                const float* ip = in + (long long)ln * si_l;
                const float* wp = w + (long long)o * Cin * 27 + kd * 9 + kh * 3 + kw;
                for (int ci = 0; ci < Cin; ++ci)
                    s += wp[(long long)ci * 27] * ip[(long long)ci * si_c];
            }
        }
    }
    if (act) s = 0.5f * s * (1.0f + erff(s * 0.70710678118f));
    out[(long long)o * L_ + l] = s;
}

// ============================================================================
// Channel attention: mean over L -> 96->48 ReLU -> 48->96 sigmoid. One block.
// ============================================================================
__global__ void attn_kernel(const float* __restrict__ y2,  // (96, L)
                            const float* __restrict__ w1, const float* __restrict__ b1,
                            const float* __restrict__ w2, const float* __restrict__ b2,
                            float* __restrict__ att)       // (96)
{
    __shared__ float mean[C_];
    __shared__ float a1[C_ / 2];
    int t = threadIdx.x;
    if (t < C_) {
        float s = 0.0f;
        const float* yp = y2 + (long long)t * L_;
        for (int l = 0; l < L_; ++l) s += yp[l];
        mean[t] = s / (float)L_;
    }
    __syncthreads();
    if (t < C_ / 2) {
        float s = b1[t];
        for (int c = 0; c < C_; ++c) s += w1[t * C_ + c] * mean[c];
        a1[t] = fmaxf(s, 0.0f);
    }
    __syncthreads();
    if (t < C_) {
        float s = b2[t];
        for (int c = 0; c < C_ / 2; ++c) s += w2[t * (C_ / 2) + c] * a1[c];
        att[t] = 1.0f / (1.0f + __expf(-s));
    }
}

// ============================================================================
// x3[l][c] = x2[l][c] * skip[c] + y2[c][l] * att[c]
// ============================================================================
__global__ void cab_res_kernel(const float* __restrict__ x2,
                               const float* __restrict__ skip,
                               const float* __restrict__ y2,
                               const float* __restrict__ att,
                               float* __restrict__ out)
{
    int i = blockIdx.x * blockDim.x + threadIdx.x;
    if (i >= L_ * C_) return;
    int l = i / C_, c = i % C_;
    out[i] = x2[i] * skip[c] + y2[(long long)c * L_ + l] * att[c];
}

// ============================================================================
// Host launcher
// ============================================================================
extern "C" void kernel_launch(void* const* d_in, const int* in_sizes, int n_in,
                              void* d_out, int out_size, void* d_ws, size_t ws_size,
                              hipStream_t stream)
{
    const float* x_in = (const float*)d_in[0];
    float* out = (float*)d_out;
    float* ws  = (float*)d_ws;

    // ---- workspace carve-up (floats, all chunk sizes multiples of 16) ----
    float* xn     = ws;                        // L*C
    float* xz     = xn     + L_ * C_;          // L*384
    float* xm_t   = xz     + L_ * 2 * DI_;     // L*Di
    float* xs_t   = xm_t   + L_ * DI_;         // 6*L*Di
    float* wpad   = xs_t   + K6_ * L_ * DI_;   // 6*48*192
    float* xdbl   = wpad   + K6_ * RP_ * DI_;  // 6*48*L
    float* delta  = xdbl   + K6_ * RP_ * L_;   // 6*L*Di
    float* ys     = delta  + K6_ * L_ * DI_;   // 6*L*Di
    float* ym     = ys     + K6_ * L_ * DI_;   // L*Di
    float* yln    = ym     + L_ * DI_;         // L*Di
    float* ssout  = yln    + L_ * DI_;         // L*C
    float* x2     = ssout  + L_ * C_;          // L*C
    float* xn2    = x2     + L_ * C_;          // L*C
    float* mid    = xn2    + L_ * C_;          // 32*L
    float* y2     = mid    + (C_ / 3) * L_;    // 96*L
    float* att    = y2     + C_ * L_;          // 96

    const int T = 256;

    for (int blk = 0; blk < 2; ++blk) {
        const int pb = 1 + blk * 25;   // params flattened in dict insertion order
        auto P = [&](int j) { return (const float*)d_in[pb + j]; };
        const float* ln1_w = P(0),  *ln1_b = P(1);
        const float* ln2_w = P(2),  *ln2_b = P(3);
        const float* skip1 = P(4),  *skip2 = P(5);
        const float* in_proj_w = P(6);
        const float* conv_w = P(7), *conv_b = P(8);
        const float* x_proj_w = P(9);
        const float* dt_w = P(10),  *dt_b = P(11);
        const float* A_logs = P(12), *Ds = P(13);
        const float* on_w = P(14),  *on_b = P(15);
        const float* out_proj_w = P(16);
        const float* c1_w = P(17),  *c1_b = P(18);
        const float* c2_w = P(19),  *c2_b = P(20);
        const float* ca1_w = P(21), *ca1_b = P(22);
        const float* ca2_w = P(23), *ca2_b = P(24);

        const float* xcur = (blk == 0) ? x_in : out;

        // 1) LN1
        ln_gate_kernel<<<(L_ + T - 1) / T, T, 0, stream>>>(
            xcur, ln1_w, ln1_b, nullptr, xn, L_, C_, C_, 0, 0, C_);

        // 2) in_proj: (1728 x 96) . (384 x 96)^T -> (1728 x 384)
        gemm_wmma_kernel<96><<<dim3(L_ / 16, (2 * DI_) / 16, 1), 32, 0, stream>>>(
            xn, in_proj_w, xz, C_, C_, 2 * DI_, 0, 0, 0);

        // 3) depthwise conv + SiLU -> (L, Di)
        dwconv_silu_kernel<<<(L_ * DI_) / T, T, 0, stream>>>(xz, conv_w, conv_b, xm_t);

        // 4) build 6 scan sequences, transposed (6, L, Di)
        build_xs_kernel<<<(K6_ * L_ * DI_) / T, T, 0, stream>>>(xm_t, xs_t);

        // 5) pad x_proj_w to (6, 48, 192); batched GEMM (48 x 192).(L x 192)^T
        pad_xproj_kernel<<<(K6_ * RP_ * DI_) / T, T, 0, stream>>>(x_proj_w, wpad);
        gemm_wmma_kernel<192><<<dim3(RP_ / 16, L_ / 16, K6_), 32, 0, stream>>>(
            wpad, xs_t, xdbl, DI_, DI_, L_,
            (long long)RP_ * DI_, (long long)L_ * DI_, (long long)RP_ * L_);

        // 6) delta = softplus(dt projection) -> (6, L, Di)
        dt_kernel<<<(K6_ * L_ * DI_) / T, T, 0, stream>>>(xdbl, dt_w, dt_b, delta);

        // 7) selective scan -> (6, L, Di)
        scan_kernel<<<(K6_ * DI_ + T - 1) / T, T, 0, stream>>>(
            xs_t, delta, xdbl, A_logs, Ds, ys);

        // 8) merge 6 directions -> (L, Di)
        merge_kernel<<<(L_ * DI_) / T, T, 0, stream>>>(ys, ym);

        // 9) out-norm with SiLU(z) gate (z = xz[:, 192:])
        ln_gate_kernel<<<(L_ + T - 1) / T, T, 0, stream>>>(
            ym, on_w, on_b, xz, yln, L_, DI_, DI_, 2 * DI_, DI_, DI_);

        // 10) out_proj: (1728 x 192) . (96 x 192)^T -> (1728 x 96)
        gemm_wmma_kernel<192><<<dim3(L_ / 16, C_ / 16, 1), 32, 0, stream>>>(
            yln, out_proj_w, ssout, DI_, DI_, C_, 0, 0, 0);

        // 11) x2 = x * skip1 + ss3d(xn)
        res_kernel<<<(L_ * C_) / T, T, 0, stream>>>(xcur, skip1, ssout, x2, L_ * C_, C_);

        // 12) LN2
        ln_gate_kernel<<<(L_ + T - 1) / T, T, 0, stream>>>(
            x2, ln2_w, ln2_b, nullptr, xn2, L_, C_, C_, 0, 0, C_);

        // 13) CAB conv1 (96 -> 32) + GELU; input layout (L, 96)
        conv3d_dense_kernel<<<((C_ / 3) * L_) / T, T, 0, stream>>>(
            xn2, c1_w, c1_b, mid, C_, C_ / 3, C_, 1, 1);

        // 14) CAB conv2 (32 -> 96); input layout (32, L) channel-major
        conv3d_dense_kernel<<<(C_ * L_) / T, T, 0, stream>>>(
            mid, c2_w, c2_b, y2, C_ / 3, C_, 1, L_, 0);

        // 15) channel attention
        attn_kernel<<<1, 128, 0, stream>>>(y2, ca1_w, ca1_b, ca2_w, ca2_b, att);

        // 16) x3 = x2 * skip2 + y2 * att  -> block output (d_out)
        cab_res_kernel<<<(L_ * C_) / T, T, 0, stream>>>(x2, skip2, y2, att, out);
    }
}
